// Attention_Lora_26792005992997
// MI455X (gfx1250) — compile-verified, matrix operand order checked by reading, not run on hardware
//
#include <hip/hip_runtime.h>

// ---------------------------------------------------------------------------
// LoRA causal attention forward for MI455X (gfx1250), bf16 WMMA everywhere.
// B=4, T=2048, C=1024, NH=16, HD=64, RANK=56 (padded to 64), scaling=8/56.
//
// Pipeline:
//   1) f32 -> bf16 conversion (+ rank padding 56->64)
//   2) xa = scaling * x @ A_attn^T                (per-wave WMMA GEMM)
//   3) qkv = x@W^T + b + xa@B^T                   (LDS double-buffered GEMM)
//      -> scatter q*1/sqrt(HD), k to [B,NH,T,HD], v transposed [B,NH,HD,T]
//   4) causal flash attention                     (LDS double-buffered K/V)
//   5) ya = scaling * y @ A_proj^T
//   6) out = y@Wp^T + b + ya@Bp^T (f32)           (LDS double-buffered GEMM)
//
// Global->LDS staging uses GLOBAL_LOAD_ASYNC_TO_LDS_B128 (ASYNCcnt) when the
// toolchain exposes the builtin, else global_load_b128 + ds_store_b128.
// ---------------------------------------------------------------------------

typedef __bf16 bf16_t;
typedef __attribute__((ext_vector_type(16))) __bf16 v16bf;
typedef __attribute__((ext_vector_type(8)))  float  v8f;

#define DEVFN __device__ __forceinline__

#if defined(__gfx1250__) &&                                        \
    __has_builtin(__builtin_amdgcn_global_load_async_to_lds_b128) && \
    __has_builtin(__builtin_amdgcn_s_wait_asynccnt)
#define USE_ASYNC_LDS 1
// Builtin signature (from hipcc diagnostic): param0 is
// "int __attribute__((vector_size(16))) __device__ *"  (addrspace(1) int4*),
// param1 the LDS-side pointer (addrspace(3)), then imm offset, imm cpol.
typedef int v4i_b128 __attribute__((vector_size(16)));
typedef __attribute__((address_space(1))) v4i_b128* gptr_b128;
typedef __attribute__((address_space(3))) v4i_b128* lptr_b128;
#else
#define USE_ASYNC_LDS 0
#endif

constexpr int   kB    = 4;
constexpr int   kT    = 2048;
constexpr int   kC    = 1024;
constexpr int   kNH   = 16;
constexpr int   kHD   = 64;
constexpr int   kRANK = 56;
constexpr int   kRPAD = 64;          // rank padded to WMMA-friendly 64
constexpr int   kM    = kB * kT;     // 8192 rows
constexpr float kSCALING = 8.0f / 56.0f;
constexpr float kQSCALE  = 0.125f;   // 1/sqrt(HD)

// ---- global->LDS 16B staging copy ----------------------------------------
DEVFN void cp_b128(const bf16_t* g, bf16_t* l) {
#if USE_ASYNC_LDS
  __builtin_amdgcn_global_load_async_to_lds_b128(
      (gptr_b128)const_cast<bf16_t*>(g), (lptr_b128)l, 0, 0);
#else
  *reinterpret_cast<uint4*>(l) = *reinterpret_cast<const uint4*>(g);
#endif
}
DEVFN void cp_wait() {
#if USE_ASYNC_LDS
  __builtin_amdgcn_s_wait_asynccnt(0);
#endif
}

// ---- fragment loaders -----------------------------------------------------
// Per CDNA5 ISA 7.12.2 (16-bit A 16x32): lane l holds row m=l%16; half=l/16
// selects K-offsets {half*8..+7} (VGPR0..3) and {16+half*8..+7} (VGPR4..7).
// Both slices are 16 contiguous bytes -> two b128 loads.
DEVFN v16bf ld_frag(const bf16_t* p) {
  v16bf f;
  reinterpret_cast<uint4*>(&f)[0] = reinterpret_cast<const uint4*>(p)[0];
  reinterpret_cast<uint4*>(&f)[1] = reinterpret_cast<const uint4*>(p + 16)[0];
  return f;
}

// LDS tile with 32-element rows, 16B chunks XOR-swizzled by ((row>>2)&3) so
// that b128 fragment loads (16 lanes, rows r..r+15) hit 16 distinct 4-bank
// groups: bank = 16*r + 4*(c ^ (r>>2))  (mod 64) -> conflict-free.
DEVFN v16bf ld_frag_sw(const bf16_t* tile, int row, int hl) {
  const int s = (row >> 2) & 3;
  const uint4* p = reinterpret_cast<const uint4*>(tile + row * 32);
  v16bf f;
  reinterpret_cast<uint4*>(&f)[0] = p[hl ^ s];
  reinterpret_cast<uint4*>(&f)[1] = p[(2 + hl) ^ s];
  return f;
}

DEVFN v8f wmma_bf16(v16bf a, v16bf b, v8f c) {
  return __builtin_amdgcn_wmma_f32_16x16x32_bf16(
      false, a, false, b, (short)0, c, false, false);
}

// ------------------------- conversion kernels ------------------------------

__global__ void cvt_f32_to_bf16(const float* __restrict__ src,
                                bf16_t* __restrict__ dst, int n) {
  int i = blockIdx.x * blockDim.x + threadIdx.x;
  if (i < n) dst[i] = (bf16_t)src[i];
}

// A-matrix [RANK, C] -> [RPAD, C] zero-padded rows, bf16
__global__ void cvt_padA(const float* __restrict__ src,
                         bf16_t* __restrict__ dst, int cols) {
  int i = blockIdx.x * blockDim.x + threadIdx.x;
  if (i >= kRPAD * cols) return;
  int rr = i / cols;
  dst[i] = (rr < kRANK) ? (bf16_t)src[i] : (bf16_t)0.0f;
}

// B-matrix [N, RANK] -> [N, RPAD] zero-padded cols, bf16
__global__ void cvt_padB(const float* __restrict__ src,
                         bf16_t* __restrict__ dst, int nrows) {
  int i = blockIdx.x * blockDim.x + threadIdx.x;
  if (i >= nrows * kRPAD) return;
  int o  = i >> 6;
  int rr = i & 63;
  dst[i] = (rr < kRANK) ? (bf16_t)src[o * kRANK + rr] : (bf16_t)0.0f;
}

// ------------------- small per-wave GEMM: xa = scal*(X @ A^T) --------------
// [M,K] x [64,K]^T -> bf16 [M,64].  One wave: 16x64 tile; block = 4 waves.
__global__ __launch_bounds__(128)
void gemm_rank(const bf16_t* __restrict__ X, const bf16_t* __restrict__ W,
               bf16_t* __restrict__ out, int K) {
  const int wave = threadIdx.x >> 5;
  const int lane = threadIdx.x & 31;
  const int hl   = lane >> 4;
  const int r    = lane & 15;
  const int m0   = (blockIdx.y * 4 + wave) * 16;

  v8f acc[4] = {};
  const bf16_t* xrow = X + (size_t)(m0 + r) * K + hl * 8;
  for (int kk = 0; kk < K; kk += 32) {
    v16bf a = ld_frag(xrow + kk);
    if (kk + 32 < K)  // global_prefetch_b8 for the next k-tile
      __builtin_prefetch(xrow + kk + 32, 0, 3);
#pragma unroll
    for (int nt = 0; nt < 4; ++nt) {
      v16bf b = ld_frag(W + (size_t)(nt * 16 + r) * K + kk + hl * 8);
      acc[nt] = wmma_bf16(a, b, acc[nt]);
    }
  }
#pragma unroll
  for (int nt = 0; nt < 4; ++nt)
#pragma unroll
    for (int v = 0; v < 8; ++v) {
      const int m = m0 + v + 8 * hl;
      out[(size_t)m * kRPAD + nt * 16 + r] = (bf16_t)(acc[nt][v] * kSCALING);
    }
}

// ------------- LDS double-buffered WMMA GEMM (modes 1=qkv, 2=proj) ---------
// Block: 256 threads (8 waves), tile 128(M) x 128(N), K staged 32 at a time.
// Wave tile: 32x64 (2 m-subtiles x 4 n-subtiles = 8 accumulators).
// Tile i+1 is staged (async when available) while tile i is consumed:
// one barrier per k-step, 64 WMMAs per 16KB staged (~64 FLOP/byte).
template <int MODE>
__global__ __launch_bounds__(256)
void gemm_tiled(const bf16_t* __restrict__ X, const bf16_t* __restrict__ W,
                const bf16_t* __restrict__ XA, const bf16_t* __restrict__ WB,
                const float* __restrict__ bias,
                bf16_t* __restrict__ o0, bf16_t* __restrict__ o1,
                bf16_t* __restrict__ o2, float* __restrict__ oF, int K) {
  __shared__ __align__(16) bf16_t lx[2][128 * 32];
  __shared__ __align__(16) bf16_t lw[2][128 * 32];

  const int t    = threadIdx.x;
  const int wave = t >> 5;
  const int lane = t & 31;
  const int hl   = lane >> 4;
  const int r    = lane & 15;
  const int wm   = wave >> 1;   // 0..3 : 32-row M slice
  const int wn   = wave & 1;    // 0..1 : 64-col N slice
  const int mBlk = blockIdx.y * 128;
  const int nBlk = blockIdx.x * 128;

  // cooperative staging: thread -> (row, 16B chunk); 2 rows per thread
  const int rowS = t >> 2;      // 0..63
  const int cS   = t & 3;

  auto stage = [&](int k0, int buf) {
#pragma unroll
    for (int j = 0; j < 2; ++j) {
      const int row = rowS + j * 64;
      const int sw  = (((row >> 2) & 3) ^ cS) * 8;
      cp_b128(X + (size_t)(mBlk + row) * K + k0 + cS * 8,
              &lx[buf][row * 32 + sw]);
      cp_b128(W + (size_t)(nBlk + row) * K + k0 + cS * 8,
              &lw[buf][row * 32 + sw]);
    }
  };

  v8f acc[2][4] = {};

  const int nsteps = K >> 5;
  stage(0, 0);
  cp_wait();
  __syncthreads();
  for (int i = 0; i < nsteps; ++i) {
    const int cur = i & 1;
    if (i + 1 < nsteps) stage((i + 1) << 5, cur ^ 1);
    const v16bf a0 = ld_frag_sw(&lx[cur][0], wm * 32 + r, hl);
    const v16bf a1 = ld_frag_sw(&lx[cur][0], wm * 32 + 16 + r, hl);
#pragma unroll
    for (int nt = 0; nt < 4; ++nt) {
      const v16bf b = ld_frag_sw(&lw[cur][0], wn * 64 + nt * 16 + r, hl);
      acc[0][nt] = wmma_bf16(a0, b, acc[0][nt]);
      acc[1][nt] = wmma_bf16(a1, b, acc[1][nt]);
    }
    cp_wait();        // next tile landed in LDS
    __syncthreads();  // all waves done reading cur / writing cur^1
  }

  // low-rank correction (K2 = 64, pre-scaled xa) straight from L2
#pragma unroll
  for (int kk = 0; kk < kRPAD; kk += 32) {
    const v16bf a0 =
        ld_frag(XA + (size_t)(mBlk + wm * 32 + r) * kRPAD + kk + hl * 8);
    const v16bf a1 =
        ld_frag(XA + (size_t)(mBlk + wm * 32 + 16 + r) * kRPAD + kk + hl * 8);
#pragma unroll
    for (int nt = 0; nt < 4; ++nt) {
      const v16bf b = ld_frag(
          WB + (size_t)(nBlk + wn * 64 + nt * 16 + r) * kRPAD + kk + hl * 8);
      acc[0][nt] = wmma_bf16(a0, b, acc[0][nt]);
      acc[1][nt] = wmma_bf16(a1, b, acc[1][nt]);
    }
  }

  // epilogue; C layout: (vgpr v, lane) -> m += v + 8*hl, n += nt*16 + r
#pragma unroll
  for (int mt = 0; mt < 2; ++mt) {
#pragma unroll
    for (int nt = 0; nt < 4; ++nt) {
#pragma unroll
      for (int v = 0; v < 8; ++v) {
        const int m = mBlk + wm * 32 + mt * 16 + v + 8 * hl;
        const int n = nBlk + wn * 64 + nt * 16 + r;
        float val = acc[mt][nt][v] + bias[n];
        if (MODE == 1) {
          const int sec = n >> 10;        // 0=q 1=k 2=v
          const int c   = n & (kC - 1);
          const int h   = c >> 6;
          const int d   = c & (kHD - 1);
          const int b_  = m >> 11;        // m / T
          const int tt  = m & (kT - 1);
          const size_t iqk = ((size_t)(b_ * kNH + h) * kT + tt) * kHD + d;
          if (sec == 0)      o0[iqk] = (bf16_t)(val * kQSCALE);
          else if (sec == 1) o1[iqk] = (bf16_t)val;
          else o2[((size_t)(b_ * kNH + h) * kHD + d) * kT + tt] = (bf16_t)val;
        } else {
          oF[(size_t)m * kC + n] = val;
        }
      }
    }
  }
}

// ------------------------- flash attention (WMMA) --------------------------
// Grid: (T/64, B*NH), block = 128 (4 waves).  All 4 waves share one (b,h);
// K/V tiles (32 keys) double-buffered in LDS, staged once per block.
// Per-wave online softmax over 16 q-rows; causal guard keeps the
// block-uniform loop convergent.
__global__ __launch_bounds__(128)
void attn_flash(const bf16_t* __restrict__ Qg, const bf16_t* __restrict__ Kg,
                const bf16_t* __restrict__ Vtg, bf16_t* __restrict__ Y) {
  __shared__ __align__(16) bf16_t lk[2][2 * 32 * 32];  // [buf][kslice][key][dim32]
  __shared__ __align__(16) bf16_t lvs[2][64 * 32];     // [buf][dim][key32]
  __shared__ __align__(16) bf16_t plds[4][16 * 32];    // per-wave P tile

  const int bh   = blockIdx.y;
  const int t    = threadIdx.x;
  const int wave = t >> 5;
  const int lane = t & 31;
  const int hl   = lane >> 4;
  const int r    = lane & 15;
  const int q0b  = blockIdx.x * 64;
  const int q0   = q0b + wave * 16;

  const bf16_t* Qp = Qg + (size_t)bh * kT * kHD;
  const bf16_t* Kp = Kg + (size_t)bh * kT * kHD;
  const bf16_t* Vp = Vtg + (size_t)bh * kHD * kT;  // [HD, T]

  v16bf qa[2];
  qa[0] = ld_frag(Qp + (size_t)(q0 + r) * kHD + 0 + hl * 8);
  qa[1] = ld_frag(Qp + (size_t)(q0 + r) * kHD + 32 + hl * 8);

  v8f oacc[4] = {};
  float mst[8], lst[8];
#pragma unroll
  for (int v = 0; v < 8; ++v) { mst[v] = -1e30f; lst[v] = 0.0f; }

  bf16_t* pl = &plds[wave][0];
  const int rowS = t >> 2;  // 0..31 (staging row)
  const int cS   = t & 3;

  auto stage_kv = [&](int kbase, int buf) {
    const int swk = ((((rowS >> 2) & 3) ^ cS)) * 8;
#pragma unroll
    for (int ks = 0; ks < 2; ++ks)
      cp_b128(Kp + (size_t)(kbase + rowS) * kHD + ks * 32 + cS * 8,
              &lk[buf][(ks * 32 + rowS) * 32 + swk]);
#pragma unroll
    for (int j = 0; j < 2; ++j) {
      const int rowd = rowS + j * 32;
      const int swv  = ((((rowd >> 2) & 3) ^ cS)) * 8;
      cp_b128(Vp + (size_t)rowd * kT + kbase + cS * 8,
              &lvs[buf][rowd * 32 + swv]);
    }
  };

  const int kend   = q0 + 16;          // this wave's causal bound (exclusive)
  const int ntiles = (q0b + 64) >> 5;  // block-uniform tile count
  stage_kv(0, 0);
  cp_wait();
  __syncthreads();
  for (int i = 0; i < ntiles; ++i) {
    const int k0  = i << 5;
    const int cur = i & 1;
    if (i + 1 < ntiles) stage_kv(k0 + 32, cur ^ 1);
    if (k0 < kend) {
      // --- S = Q * K^T for 32 keys (two 16-col n-tiles) ---
      v8f s0 = {}, s1 = {};
#pragma unroll
      for (int kk = 0; kk < 2; ++kk) {
        const v16bf kf0 = ld_frag_sw(&lk[cur][kk * 32 * 32], r, hl);
        const v16bf kf1 = ld_frag_sw(&lk[cur][kk * 32 * 32], 16 + r, hl);
        s0 = wmma_bf16(qa[kk], kf0, s0);
        s1 = wmma_bf16(qa[kk], kf1, s1);
      }
      // --- causal mask + online softmax (row = (v,hl), col = r) ---
#pragma unroll
      for (int v = 0; v < 8; ++v) {
        const int m = q0 + v + 8 * hl;
        if (k0 + r > m)      s0[v] = -3.0e38f;
        if (k0 + 16 + r > m) s1[v] = -3.0e38f;
        float mx = fmaxf(s0[v], s1[v]);
        mx = fmaxf(mx, __shfl_xor(mx, 1));
        mx = fmaxf(mx, __shfl_xor(mx, 2));
        mx = fmaxf(mx, __shfl_xor(mx, 4));
        mx = fmaxf(mx, __shfl_xor(mx, 8));
        const float mnew = fmaxf(mst[v], mx);
        const float corr = __expf(mst[v] - mnew);
        const float p0 = __expf(s0[v] - mnew);
        const float p1 = __expf(s1[v] - mnew);
        float rs = p0 + p1;
        rs += __shfl_xor(rs, 1);
        rs += __shfl_xor(rs, 2);
        rs += __shfl_xor(rs, 4);
        rs += __shfl_xor(rs, 8);
        lst[v] = lst[v] * corr + rs;
        mst[v] = mnew;
        oacc[0][v] *= corr;
        oacc[1][v] *= corr;
        oacc[2][v] *= corr;
        oacc[3][v] *= corr;
        // stage P (C-layout) into LDS, chunk-swizzled like ld_frag_sw expects
        const int row = v + 8 * hl;
        const int s   = (row >> 2) & 3;
        pl[row * 32 + (((r >> 3) ^ s) << 3) + (r & 7)]       = (bf16_t)p0;
        pl[row * 32 + ((((r >> 3) + 2) ^ s) << 3) + (r & 7)] = (bf16_t)p1;
      }
      // --- O += P(16x32) * V(32x64) ---
      const v16bf pfrag = ld_frag_sw(pl, r, hl);
#pragma unroll
      for (int dt = 0; dt < 4; ++dt) {
        const v16bf vf = ld_frag_sw(&lvs[cur][0], dt * 16 + r, hl);
        oacc[dt] = wmma_bf16(pfrag, vf, oacc[dt]);
      }
    }
    cp_wait();
    __syncthreads();
  }

  // epilogue: normalize rows, write y bf16 [B, T, C] (head-interleaved)
  const int b_ = bh / kNH;
  const int h  = bh % kNH;
#pragma unroll
  for (int dt = 0; dt < 4; ++dt) {
#pragma unroll
    for (int v = 0; v < 8; ++v) {
      const int m = q0 + v + 8 * hl;
      const float val = oacc[dt][v] / lst[v];
      Y[((size_t)b_ * kT + m) * kC + h * kHD + dt * 16 + r] = (bf16_t)val;
    }
  }
}

// ------------------------------ host side ----------------------------------

extern "C" void kernel_launch(void* const* d_in, const int* in_sizes, int n_in,
                              void* d_out, int out_size, void* d_ws,
                              size_t ws_size, hipStream_t stream) {
  (void)in_sizes; (void)n_in; (void)out_size; (void)ws_size;
  const float* x      = (const float*)d_in[0];
  const float* W_attn = (const float*)d_in[1];
  const float* b_attn = (const float*)d_in[2];
  const float* A_attn = (const float*)d_in[3];
  const float* B_attn = (const float*)d_in[4];
  const float* W_proj = (const float*)d_in[5];
  const float* b_proj = (const float*)d_in[6];
  const float* A_proj = (const float*)d_in[7];
  const float* B_proj = (const float*)d_in[8];
  float* out = (float*)d_out;

  // workspace carve (~96 MB)
  char* ws = (char*)d_ws;
  size_t off = 0;
  auto carve = [&](size_t elems) {
    bf16_t* p = (bf16_t*)(ws + off);
    off += (elems * sizeof(bf16_t) + 255) & ~(size_t)255;
    return p;
  };
  bf16_t* xbf   = carve((size_t)kM * kC);
  bf16_t* wa_bf = carve((size_t)3 * kC * kC);
  bf16_t* aa_bf = carve((size_t)kRPAD * kC);
  bf16_t* ba_bf = carve((size_t)3 * kC * kRPAD);
  bf16_t* wp_bf = carve((size_t)kC * kC);
  bf16_t* ap_bf = carve((size_t)kRPAD * kC);
  bf16_t* bp_bf = carve((size_t)kC * kRPAD);
  bf16_t* xa    = carve((size_t)kM * kRPAD);   // x @ A_attn^T (pre-scaled)
  bf16_t* qbf   = carve((size_t)kM * kC);      // [B,NH,T,HD], *1/sqrt(HD)
  bf16_t* kbf   = carve((size_t)kM * kC);      // [B,NH,T,HD]
  bf16_t* vtbf  = carve((size_t)kM * kC);      // [B,NH,HD,T]
  bf16_t* ybf   = carve((size_t)kM * kC);      // attention out [B,T,C]
  bf16_t* ya    = carve((size_t)kM * kRPAD);   // y @ A_proj^T (pre-scaled)

  const int cvtB = 256;
  auto cvtG = [](int n) { return (n + 255) / 256; };

  // 1) precision conversion + rank padding
  cvt_f32_to_bf16<<<cvtG(kM * kC), cvtB, 0, stream>>>(x, xbf, kM * kC);
  cvt_f32_to_bf16<<<cvtG(3 * kC * kC), cvtB, 0, stream>>>(W_attn, wa_bf,
                                                          3 * kC * kC);
  cvt_f32_to_bf16<<<cvtG(kC * kC), cvtB, 0, stream>>>(W_proj, wp_bf, kC * kC);
  cvt_padA<<<cvtG(kRPAD * kC), cvtB, 0, stream>>>(A_attn, aa_bf, kC);
  cvt_padA<<<cvtG(kRPAD * kC), cvtB, 0, stream>>>(A_proj, ap_bf, kC);
  cvt_padB<<<cvtG(3 * kC * kRPAD), cvtB, 0, stream>>>(B_attn, ba_bf, 3 * kC);
  cvt_padB<<<cvtG(kC * kRPAD), cvtB, 0, stream>>>(B_proj, bp_bf, kC);

  // 2) xa = scaling * (x @ A_attn^T)   [8192, 64]
  gemm_rank<<<dim3(1, kM / 64), dim3(128), 0, stream>>>(xbf, aa_bf, xa, kC);
  // 3) qkv = x@W^T + b + xa@B^T  -> scatter q/k/vt
  gemm_tiled<1><<<dim3(3 * kC / 128, kM / 128), dim3(256), 0, stream>>>(
      xbf, wa_bf, xa, ba_bf, b_attn, qbf, kbf, vtbf, nullptr, kC);
  // 4) causal flash attention
  attn_flash<<<dim3(kT / 64, kB * kNH), dim3(128), 0, stream>>>(qbf, kbf, vtbf,
                                                                ybf);
  // 5) ya = scaling * (y @ A_proj^T)
  gemm_rank<<<dim3(1, kM / 64), dim3(128), 0, stream>>>(ybf, ap_bf, ya, kC);
  // 6) out = y@Wp^T + b + ya@Bp^T   (f32)
  gemm_tiled<2><<<dim3(kC / 128, kM / 128), dim3(256), 0, stream>>>(
      ybf, wp_bf, ya, bp_bf, b_proj, nullptr, nullptr, nullptr, out, kC);
}
